// DecoderAttention_13322988552310
// MI455X (gfx1250) — compile-verified, matrix operand order checked by reading, not run on hardware
//
#include <hip/hip_runtime.h>
#include <hip/hip_bf16.h>
#include <stdint.h>

// ---------------------------------------------------------------------------
// Problem constants (match reference)
// ---------------------------------------------------------------------------
#define B_   32
#define T_   64
#define E_   300
#define Z_   1100
#define H_   800
#define V_   32000
#define IN_    (E_ + Z_)      // 1400
#define INP_   1408           // IN padded to multiple of 32 (bf16 K-dim)
#define FOURH  (4 * H_)       // 3200
#define TWOH   (2 * H_)       // 1600

typedef __attribute__((ext_vector_type(16))) __bf16       v16bf;
typedef __attribute__((ext_vector_type(8)))  float        v8f;
typedef __attribute__((ext_vector_type(4)))  unsigned int u32x4;
typedef __attribute__((ext_vector_type(8)))  int          i32x8;
typedef __attribute__((ext_vector_type(4)))  int          i32x4;

__device__ __forceinline__ float sigmoidf_(float x) {
    return 1.0f / (1.0f + __expf(-x));
}

__device__ __forceinline__ v8f wmma_bf16(v16bf a, v16bf b, v8f c) {
    return __builtin_amdgcn_wmma_f32_16x16x32_bf16(
               false, a, false, b, (short)0, c, false, false);
}

// ---------------------------------------------------------------------------
// Software-pipelined 4-accumulator WMMA pass:
//   acc[a] += A[arow0:+16, :K] @ W[wrow0+a*wrowStride:+16, :K]^T
// Per-lane pointers walk the K dimension (+32 elems/step); the next K-step's
// fragments are loaded into registers while the current WMMAs execute, so the
// compiler can overlap loads with matrix ops (non-zero loadcnt waits).
// Requires K % 32 == 0, K >= 32, ld % 16 == 0, 32B-aligned rows.
// ---------------------------------------------------------------------------
__device__ __forceinline__ void wmma_pass4(v8f* acc,
                                           const __bf16* __restrict__ A,
                                           int lda, int arow0,
                                           const __bf16* __restrict__ W,
                                           int ldw, int wrow0, int wrowStride,
                                           int K, int lane) {
    const int col = lane & 15;
    const int kh  = (lane >> 4) << 4;
    const __bf16* pA = A + (size_t)(arow0 + col) * lda + kh;
    const __bf16* pW[4];
#pragma unroll
    for (int a = 0; a < 4; ++a)
        pW[a] = W + (size_t)(wrow0 + a * wrowStride + col) * ldw + kh;

    v16bf af = *(const v16bf*)pA;
    v16bf wf[4];
#pragma unroll
    for (int a = 0; a < 4; ++a) wf[a] = *(const v16bf*)pW[a];

    for (int k0 = 32; k0 < K; k0 += 32) {
        pA += 32;
        v16bf afn = *(const v16bf*)pA;
        v16bf wfn[4];
#pragma unroll
        for (int a = 0; a < 4; ++a) {
            pW[a] += 32;
            wfn[a] = *(const v16bf*)pW[a];
        }
        __builtin_prefetch(pA + 32, 0, 1);
        __builtin_prefetch(pW[0] + 32, 0, 1);
#pragma unroll
        for (int a = 0; a < 4; ++a) acc[a] = wmma_bf16(af, wf[a], acc[a]);
        af = afn;
#pragma unroll
        for (int a = 0; a < 4; ++a) wf[a] = wfn[a];
    }
#pragma unroll
    for (int a = 0; a < 4; ++a) acc[a] = wmma_bf16(af, wf[a], acc[a]);
}

// ---------------------------------------------------------------------------
// fp32 -> bf16 conversion with optional K padding (dst[r, k>=srcK] = 0)
// ---------------------------------------------------------------------------
__global__ void cvt_f32_bf16_pad(const float* __restrict__ src, int srcK,
                                 __bf16* __restrict__ dst, int dstK,
                                 long long total) {
    long long i = (long long)blockIdx.x * blockDim.x + threadIdx.x;
    if (i >= total) return;
    int k = (int)(i % dstK);
    long long r = i / dstK;
    dst[i] = (k < srcK) ? (__bf16)src[r * (long long)srcK + k] : (__bf16)0.0f;
}

// x_seq_bf16[t][b][k] : [T,B,INP_] padded-bf16 concat(dec, z)
__global__ void concat_xz_bf16(const float* __restrict__ dec,
                               const float* __restrict__ z,
                               __bf16* __restrict__ x_seq) {
    const long long total = (long long)T_ * B_ * INP_;
    long long idx = (long long)blockIdx.x * blockDim.x + threadIdx.x;
    if (idx >= total) return;
    int k = (int)(idx % INP_);
    long long r = idx / INP_;
    int b = (int)(r % B_);
    int t = (int)(r / B_);
    float v = 0.0f;
    if (k < E_)        v = dec[((size_t)b * T_ + t) * E_ + k];
    else if (k < IN_)  v = z[(size_t)b * Z_ + (k - E_)];
    x_seq[idx] = (__bf16)v;
}

// ---------------------------------------------------------------------------
// Generic GEMM: C(f32) = A(bf16) @ W(bf16)^T + bias.  One wave -> 16x64 strip
// (4 accumulators share one A fragment).  blockDim.x = 128 (4 waves).
// Requires M%16==0, N%64==0, K%32==0.
// ---------------------------------------------------------------------------
__global__ void gemm_w64_wmma(const __bf16* __restrict__ A, int lda,
                              const __bf16* __restrict__ W, int ldw,
                              const float* __restrict__ bias,
                              float* __restrict__ C, int ldc,
                              int M, int N, int K) {
    const int lane = threadIdx.x & 31;
    const int wid  = threadIdx.x >> 5;
    const int nchunks = N >> 6;
    const int wave = blockIdx.x * 4 + wid;
    if (wave >= (M >> 4) * nchunks) return;     // uniform per wave
    const int mt = wave / nchunks;
    const int nc = wave - mt * nchunks;
    const int m0 = mt << 4;
    const int n0 = nc << 6;
    const int col = lane & 15;

    v8f acc[4];
#pragma unroll
    for (int a = 0; a < 4; ++a) {
        const float bv = bias ? bias[n0 + a * 16 + col] : 0.0f;
#pragma unroll
        for (int r = 0; r < 8; ++r) acc[a][r] = bv;
    }

    wmma_pass4(acc, A, lda, m0, W, ldw, n0, 16, K, lane);

    const int mbase = m0 + ((lane >> 4) << 3);
#pragma unroll
    for (int a = 0; a < 4; ++a)
#pragma unroll
        for (int r = 0; r < 8; ++r)
            C[(size_t)(mbase + r) * ldc + n0 + a * 16 + col] = acc[a][r];
}

// ---------------------------------------------------------------------------
// Logits GEMM with pipelined TDM-staged A tiles (double-buffered LDS).
// grid = (N/256, M/16), blockDim = 128 (4 waves). Wave w covers n0 = bx*256+w*64.
// The Tensor Data Mover DMAs K-step i+1's 16x32 bf16 A tile into the alternate
// LDS buffer while the waves compute on K-step i (TDM ops of one wave complete
// in order => s_wait_tensorcnt(1) guarantees the older transfer has landed).
// ---------------------------------------------------------------------------
__global__ void gemm_w64_tdm(const __bf16* __restrict__ A, int lda,
                             const __bf16* __restrict__ W, int ldw,
                             const float* __restrict__ bias,
                             float* __restrict__ C, int ldc, int K) {
    __shared__ __attribute__((aligned(64))) __bf16 sA[2][16 * 32];
    const int lane = threadIdx.x & 31;
    const int wid  = threadIdx.x >> 5;
    const int m0 = blockIdx.y << 4;
    const int n0 = (blockIdx.x << 8) + (wid << 6);
    const int col = lane & 15;
    const int kh  = (lane >> 4) << 4;

    v8f acc[4];
#pragma unroll
    for (int a = 0; a < 4; ++a) {
        const float bv = bias ? bias[n0 + a * 16 + col] : 0.0f;
#pragma unroll
        for (int r = 0; r < 8; ++r) acc[a][r] = bv;
    }

    const __bf16* pW[4];
#pragma unroll
    for (int a = 0; a < 4; ++a)
        pW[a] = W + (size_t)(n0 + a * 16 + col) * ldw + kh;

#if __has_builtin(__builtin_amdgcn_tensor_load_to_lds) && \
    __has_builtin(__builtin_amdgcn_s_wait_tensorcnt)
    // D# builder: 2D tile, 16 rows x 32 cols of 2-byte elements,
    // row stride = lda elements, dest = LDS buffer `buf`.
    auto issue_tdm = [&](int buf, int k0) {
        unsigned long long ga =
            (unsigned long long)(uintptr_t)(A + (size_t)m0 * lda + k0);
        u32x4 g0;
        g0[0] = 1u;                                   // count=1 (valid D#)
        g0[1] = (unsigned)(uintptr_t)&sA[buf][0];     // lds_addr (bytes)
        g0[2] = (unsigned)(ga & 0xFFFFFFFFu);         // global_addr lo
        g0[3] = (unsigned)((ga >> 32) & 0x1FFFFFFu)   // global_addr hi
              | (2u << 30);                           // type=2 ("image")
        i32x8 g1;
        g1[0] = 1 << 16;                              // data_size=1 => 2B
        g1[1] = (int)(((unsigned)K & 0xFFFFu) << 16);         // dim0 lo16
        g1[2] = (int)((((unsigned)K >> 16) & 0xFFFFu)         // dim0 hi16
              | (16u << 16));                                 // dim1 = 16
        g1[3] = (int)(32u << 16);                     // tile_dim0 = 32
        g1[4] = 16;                                   // tile_dim1 = 16
        g1[5] = (int)(unsigned)lda;                   // dim0_stride lo32
        g1[6] = 0;
        g1[7] = 0;
        i32x4 gz4 = (i32x4){0, 0, 0, 0};              // groups 2/3 unused (2D)
        i32x8 gz8 = (i32x8){0, 0, 0, 0, 0, 0, 0, 0};  // aux group (unused)
        __builtin_amdgcn_tensor_load_to_lds(g0, g1, gz4, gz4, gz8, 0);
    };
    if (wid == 0) issue_tdm(0, 0);
    int buf = 0;
    for (int k0 = 0; k0 < K; k0 += 32) {
        if (wid == 0) {
            if (k0 + 32 < K) {
                issue_tdm(buf ^ 1, k0 + 32);              // prefetch next tile
                __builtin_amdgcn_s_wait_tensorcnt(1);     // oldest has landed
            } else {
                __builtin_amdgcn_s_wait_tensorcnt(0);
            }
        }
        __syncthreads();                  // sA[buf] valid for all waves
        v16bf af = *(const v16bf*)(&sA[buf][(lane & 15) * 32 + kh]);
        v16bf wf[4];
#pragma unroll
        for (int a = 0; a < 4; ++a) {
            wf[a] = *(const v16bf*)pW[a];
            pW[a] += 32;
        }
#pragma unroll
        for (int a = 0; a < 4; ++a) acc[a] = wmma_bf16(af, wf[a], acc[a]);
        __syncthreads();                  // reads done before buf reuse
        buf ^= 1;
    }
#else
    for (int k0 = 0; k0 < K; k0 += 32) {
        for (int i = threadIdx.x; i < 16 * 32; i += blockDim.x) {
            int r = i >> 5, c = i & 31;
            sA[0][i] = A[(size_t)(m0 + r) * lda + k0 + c];
        }
        __syncthreads();
        v16bf af = *(const v16bf*)(&sA[0][(lane & 15) * 32 + kh]);
        v16bf wf[4];
#pragma unroll
        for (int a = 0; a < 4; ++a) {
            wf[a] = *(const v16bf*)pW[a];
            pW[a] += 32;
        }
#pragma unroll
        for (int a = 0; a < 4; ++a) acc[a] = wmma_bf16(af, wf[a], acc[a]);
        __syncthreads();
    }
#endif

    const int mbase = m0 + ((lane >> 4) << 3);
#pragma unroll
    for (int a = 0; a < 4; ++a)
#pragma unroll
        for (int r = 0; r < 8; ++r)
            C[(size_t)(mbase + r) * ldc + n0 + a * 16 + col] = acc[a][r];
}

// ---------------------------------------------------------------------------
// Fused LSTM cell step with WMMA.  Wave owns (m-tile, j-tile) with 4 gate
// accumulators (i,f,g,o at column offsets 0,H,2H,3H).  Seeds from preG
// (precomputed x-gates+bias, f32) or bias; accumulates optional x@W_ih^T and
// h_prev@W_hh^T (all bf16, software-pipelined); writes h in f32+bf16, c f32.
// blockDim.x = 128.
// ---------------------------------------------------------------------------
__global__ void lstm_cell_wmma(const float* __restrict__ preG, int preG_ld,
                               const __bf16* __restrict__ x, int Kx,
                               const __bf16* __restrict__ W_ih,
                               const __bf16* __restrict__ hprev,
                               const __bf16* __restrict__ W_hh,
                               const float* __restrict__ bias,
                               const float* __restrict__ cprev,
                               float* __restrict__ hnew,
                               __bf16* __restrict__ hnew_bf,
                               float* __restrict__ cnew) {
    const int lane = threadIdx.x & 31;
    const int wid  = threadIdx.x >> 5;
    const int jtiles = H_ >> 4;                 // 50
    const int tile = blockIdx.x * 4 + wid;
    if (tile >= (B_ >> 4) * jtiles) return;     // uniform per wave
    const int mt = tile / jtiles;
    const int jt = tile - mt * jtiles;
    const int m0 = mt << 4;
    const int j0 = jt << 4;
    const int col = lane & 15;
    const int hi  = lane >> 4;

    v8f acc[4];
    if (preG) {
#pragma unroll
        for (int g = 0; g < 4; ++g) {
            const int nOff = g * H_ + j0 + col;
#pragma unroll
            for (int r = 0; r < 8; ++r)
                acc[g][r] = preG[(size_t)(m0 + r + hi * 8) * preG_ld + nOff];
        }
    } else {
#pragma unroll
        for (int g = 0; g < 4; ++g) {
            const float bv = bias ? bias[g * H_ + j0 + col] : 0.0f;
#pragma unroll
            for (int r = 0; r < 8; ++r) acc[g][r] = bv;
        }
    }

    if (x)  // layer-1 input pass: gates += x @ W_ih^T
        wmma_pass4(acc, x, Kx, m0, W_ih, Kx, j0, H_, Kx, lane);

    // recurrent pass: gates += h_prev @ W_hh^T (K = H)
    wmma_pass4(acc, hprev, H_, m0, W_hh, H_, j0, H_, H_, lane);

    // elementwise cell update (f32)
#pragma unroll
    for (int r = 0; r < 8; ++r) {
        const int m = m0 + r + hi * 8;
        const int j = j0 + col;
        const size_t off = (size_t)m * H_ + j;
        float iv = sigmoidf_(acc[0][r]);
        float fv = sigmoidf_(acc[1][r]);
        float gv = tanhf(acc[2][r]);
        float ov = sigmoidf_(acc[3][r]);
        float cn = fv * cprev[off] + iv * gv;
        float hn = ov * tanhf(cn);
        cnew[off]    = cn;
        hnew[off]    = hn;
        hnew_bf[off] = (__bf16)hn;
    }
}

// ---------------------------------------------------------------------------
// Dot-product attention + concat(h_top, context) -> outs (bf16 for the
// logits GEMM).  grid = B_, blockDim = 256.  Row index = b*T + t.
// ---------------------------------------------------------------------------
__global__ void attention_kernel(const float* __restrict__ h_top,   // [B,H] f32
                                 const float* __restrict__ enc,     // [B,T,H]
                                 __bf16* __restrict__ outs,         // [B,T,2H]
                                 int t) {
    const int b = blockIdx.x;
    const int tid = threadIdx.x;
    __shared__ float sc[T_];
    __shared__ float part[256];

    const float* hb = h_top + (size_t)b * H_;
    const float* eb = enc + (size_t)b * T_ * H_;

    const int s = tid >> 2, sub = tid & 3;   // 64 scores x 4 partial threads
    float p = 0.0f;
    for (int j = sub; j < H_; j += 4) p += hb[j] * eb[(size_t)s * H_ + j];
    part[tid] = p;
    __syncthreads();
    if (tid < T_)
        sc[tid] = part[tid * 4] + part[tid * 4 + 1] +
                  part[tid * 4 + 2] + part[tid * 4 + 3];
    __syncthreads();
    if (tid == 0) {   // tiny softmax over 64
        float mx = sc[0];
        for (int i = 1; i < T_; ++i) mx = fmaxf(mx, sc[i]);
        float sum = 0.0f;
        for (int i = 0; i < T_; ++i) { float e = __expf(sc[i] - mx); sc[i] = e; sum += e; }
        float inv = 1.0f / sum;
        for (int i = 0; i < T_; ++i) sc[i] *= inv;
    }
    __syncthreads();

    __bf16* orow = outs + ((size_t)b * T_ + t) * TWOH;
    for (int j = tid; j < H_; j += 256) {
        float ctx = 0.0f;
        for (int s2 = 0; s2 < T_; ++s2) ctx += sc[s2] * eb[(size_t)s2 * H_ + j];
        orow[j]      = (__bf16)hb[j];
        orow[H_ + j] = (__bf16)ctx;
    }
}

// ---------------------------------------------------------------------------
// Small helpers
// ---------------------------------------------------------------------------
__global__ void zero_kernel(float* __restrict__ p, int n) {
    int i = blockIdx.x * blockDim.x + threadIdx.x;
    if (i < n) p[i] = 0.0f;
}

__global__ void copy_final_kernel(const float* __restrict__ h0f,
                                  const float* __restrict__ h1f,
                                  const float* __restrict__ c0f,
                                  const float* __restrict__ c1f,
                                  float* __restrict__ out) {
    const int n = B_ * H_;   // 25600
    int i = blockIdx.x * blockDim.x + threadIdx.x;
    if (i >= n) return;
    out[i]         = h0f[i];
    out[n + i]     = h1f[i];
    out[2 * n + i] = c0f[i];
    out[3 * n + i] = c1f[i];
}

// ---------------------------------------------------------------------------
// kernel_launch
// ---------------------------------------------------------------------------
extern "C" void kernel_launch(void* const* d_in, const int* in_sizes, int n_in,
                              void* d_out, int out_size, void* d_ws, size_t ws_size,
                              hipStream_t stream) {
    (void)in_sizes; (void)n_in; (void)out_size; (void)ws_size;

    const float* dec   = (const float*)d_in[0];
    const float* z     = (const float*)d_in[1];
    const float* enc   = (const float*)d_in[2];
    const float* W_ih0 = (const float*)d_in[4];
    const float* W_hh0 = (const float*)d_in[5];
    const float* b0    = (const float*)d_in[6];
    const float* W_ih1 = (const float*)d_in[7];
    const float* W_hh1 = (const float*)d_in[8];
    const float* b1    = (const float*)d_in[9];
    const float* W_out = (const float*)d_in[10];
    const float* b_out = (const float*)d_in[11];

    float* logits = (float*)d_out;                       // [B,T,V]
    float* hcf    = logits + (size_t)B_ * T_ * V_;       // h_f then c_f

    // -------- workspace carve-out (256B aligned chunks) --------
    char* wsb = (char*)d_ws;
    auto carve = [&](size_t bytes) -> char* {
        char* p = wsb;
        wsb += (bytes + 255) & ~(size_t)255;
        return p;
    };
    __bf16* x_seqb = (__bf16*)carve((size_t)T_ * B_ * INP_ * 2);     //  5.8 MB
    __bf16* Wih0b  = (__bf16*)carve((size_t)FOURH * INP_ * 2);       //  9.0 MB
    __bf16* Whh0b  = (__bf16*)carve((size_t)FOURH * H_ * 2);         //  5.1 MB
    __bf16* Wih1b  = (__bf16*)carve((size_t)FOURH * H_ * 2);         //  5.1 MB
    __bf16* Whh1b  = (__bf16*)carve((size_t)FOURH * H_ * 2);         //  5.1 MB
    __bf16* Woutb  = (__bf16*)carve((size_t)V_ * TWOH * 2);          // 102.4 MB
    float*  G0pre  = (float*)carve((size_t)T_ * B_ * FOURH * 4);     // 26.2 MB
    float*  h0buf  = (float*)carve(2 * (size_t)B_ * H_ * 4);
    float*  c0buf  = (float*)carve(2 * (size_t)B_ * H_ * 4);
    float*  h1buf  = (float*)carve(2 * (size_t)B_ * H_ * 4);
    float*  c1buf  = (float*)carve(2 * (size_t)B_ * H_ * 4);
    __bf16* h0bb   = (__bf16*)carve(2 * (size_t)B_ * H_ * 2);
    __bf16* h1bb   = (__bf16*)carve(2 * (size_t)B_ * H_ * 2);
    __bf16* outsb  = (__bf16*)carve((size_t)B_ * T_ * TWOH * 2);     //  6.6 MB

    const int BH = B_ * H_;

    // 1) build padded-bf16 x_seq = concat(dec, z) in [T,B,INP] layout
    {
        long long total = (long long)T_ * B_ * INP_;
        concat_xz_bf16<<<(unsigned)((total + 255) / 256), 256, 0, stream>>>(
            dec, z, x_seqb);
    }

    // 2) convert weights fp32 -> bf16 (W_ih0 padded 1400->1408)
    {
        long long t0 = (long long)FOURH * INP_;
        cvt_f32_bf16_pad<<<(unsigned)((t0 + 255) / 256), 256, 0, stream>>>(
            W_ih0, IN_, Wih0b, INP_, t0);
        long long t1 = (long long)FOURH * H_;
        cvt_f32_bf16_pad<<<(unsigned)((t1 + 255) / 256), 256, 0, stream>>>(
            W_hh0, H_, Whh0b, H_, t1);
        cvt_f32_bf16_pad<<<(unsigned)((t1 + 255) / 256), 256, 0, stream>>>(
            W_ih1, H_, Wih1b, H_, t1);
        cvt_f32_bf16_pad<<<(unsigned)((t1 + 255) / 256), 256, 0, stream>>>(
            W_hh1, H_, Whh1b, H_, t1);
        long long t2 = (long long)V_ * TWOH;
        cvt_f32_bf16_pad<<<(unsigned)((t2 + 255) / 256), 256, 0, stream>>>(
            W_out, TWOH, Woutb, TWOH, t2);
    }

    // 3) batched layer-0 input gates for ALL timesteps:
    //    G0pre = x_seq @ W_ih0^T + b0       [2048,1408]x[1408,3200]
    {
        int waves = ((T_ * B_) >> 4) * (FOURH >> 6);   // 128*50 = 6400
        gemm_w64_wmma<<<waves / 4, 128, 0, stream>>>(
            x_seqb, INP_, Wih0b, INP_, b0, G0pre, FOURH, T_ * B_, FOURH, INP_);
    }

    // 4) zero-init state buffer 0 (f32 h/c and bf16 h mirrors)
    zero_kernel<<<(BH + 255) / 256, 256, 0, stream>>>(h0buf, BH);
    zero_kernel<<<(BH + 255) / 256, 256, 0, stream>>>(c0buf, BH);
    zero_kernel<<<(BH + 255) / 256, 256, 0, stream>>>(h1buf, BH);
    zero_kernel<<<(BH + 255) / 256, 256, 0, stream>>>(c1buf, BH);
    zero_kernel<<<(BH / 2 + 255) / 256, 256, 0, stream>>>((float*)h0bb, BH / 2);
    zero_kernel<<<(BH / 2 + 255) / 256, 256, 0, stream>>>((float*)h1bb, BH / 2);

    // 5) sequential scan over T with double-buffered state
    const int cellBlocks = ((B_ >> 4) * (H_ >> 4)) / 4;  // 25
    for (int t = 0; t < T_; ++t) {
        const int cur = t & 1, nxt = cur ^ 1;
        // layer 0: seeded from precomputed gates, recurrent-only WMMA pass
        lstm_cell_wmma<<<cellBlocks, 128, 0, stream>>>(
            G0pre + (size_t)t * B_ * FOURH, FOURH,
            nullptr, 0, nullptr,
            h0bb + cur * BH, Whh0b, nullptr,
            c0buf + cur * BH,
            h0buf + nxt * BH, h0bb + nxt * BH, c0buf + nxt * BH);
        // layer 1: input pass (h0_new @ W_ih1^T) + recurrent pass
        lstm_cell_wmma<<<cellBlocks, 128, 0, stream>>>(
            nullptr, 0,
            h0bb + nxt * BH, H_, Wih1b,
            h1bb + cur * BH, Whh1b, b1,
            c1buf + cur * BH,
            h1buf + nxt * BH, h1bb + nxt * BH, c1buf + nxt * BH);
        // attention + concat -> outs (bf16)
        attention_kernel<<<B_, 256, 0, stream>>>(
            h1buf + nxt * BH, enc, outsb, t);
    }

    // 6) logits = outs @ W_out^T + b_out   [2048,1600]x[1600,32000]
    //    pipelined TDM-staged shared A tiles; W_out(bf16, 102MB) L2-resident.
    {
        dim3 grid(V_ / 256, (B_ * T_) / 16);   // (125, 128)
        gemm_w64_tdm<<<grid, 128, 0, stream>>>(
            outsb, TWOH, Woutb, TWOH, b_out, logits, V_, TWOH);
    }

    // 7) final h/c -> tail of d_out (after 64 steps the live buffer is idx 0)
    copy_final_kernel<<<(BH + 255) / 256, 256, 0, stream>>>(
        h0buf, h1buf, c0buf, c1buf, hcf);
}
